// EMRA_focus_64802466562055
// MI455X (gfx1250) — matrix operand order repaired and checked by reading, not compiled, not run on hardware
//
#include <hip/hip_runtime.h>
#include <hip/hip_bf16.h>

typedef __attribute__((ext_vector_type(16))) _Float16 v16h;
typedef __attribute__((ext_vector_type(8)))  float    v8f;
typedef __attribute__((ext_vector_type(4)))  unsigned int v4u;
typedef __attribute__((ext_vector_type(8)))  int      v8i;
typedef __attribute__((ext_vector_type(4)))  int      v4i;

#define BB   16
#define CC   64
#define HH   144
#define WW   144
#define HS   48          // downsampled spatial
#define SP   2304        // 48*48
#define NTOT 36864       // BB*SP  (GEMM N)
#define NT   2304        // NTOT/16 n-tiles
#define PIX  20736       // 144*144

// ---------------------------------------------------------------------------
// Stage A: maxpool3x3(pad1) + focus downsample, emit f16 packed directly in
// WMMA B-fragment layout (16x16x32 f16): fragment(g, ntile, kstep) is 512
// halves; lane L = hi*16 + nlo holds 16 contiguous halves e=0..15 with
// K = hi*16 + e (channel = kstep*32 + K), N = ntile*16 + nlo.
// ---------------------------------------------------------------------------
__global__ __launch_bounds__(256) void k_pool_focus(const float* __restrict__ x,
                                                    _Float16* __restrict__ xf) {
  int idx = blockIdx.x * 256 + threadIdx.x;
  if (idx >= BB * CC * PIX) return;
  int p  = idx % PIX;
  int bc = idx / PIX;
  int c  = bc % CC, b = bc / CC;
  int yy = p / WW, xx = p % WW;
  const float* plane = x + (size_t)bc * PIX;
  int y0 = yy > 0 ? yy - 1 : 0, y1 = yy < HH - 1 ? yy + 1 : HH - 1;
  int x0 = xx > 0 ? xx - 1 : 0, x1 = xx < WW - 1 ? xx + 1 : WW - 1;
  float m = -3.4e38f;
  for (int ry = y0; ry <= y1; ++ry)
    for (int rx = x0; rx <= x1; ++rx) {
      float v = plane[ry * WW + rx];
      m = v > m ? v : m;
    }
  int i = yy % 3, j = xx % 3, hh = yy / 3, ww = xx / 3;
  int g = i * 3 + j;
  int n = b * SP + hh * HS + ww;
  int s = c >> 5, K = c & 31, hi = K >> 4, e = K & 15;
  int nt = n >> 4, nlo = n & 15, L = hi * 16 + nlo;
  size_t addr = ((((size_t)(g * NT + nt)) * 2 + s) * 32 + L) * 16 + e;
  xf[addr] = (_Float16)m;
}

// ---------------------------------------------------------------------------
// Stage B: 9 grouped GEMMs  y[g*64+o, n] = sum_i w[g,o,i] * xf[g,i,n]
// M=64 (4 tiles), K=64 (2 steps of 32), N=36864 (2304 tiles of 16).
// Block = 256 thr = 8 waves; wave -> one n-tile, 4 m-tiles, 8 WMMAs.
// A staged in LDS (contiguous K = hi*16 + e, same order as B).
// B staged in LDS by the Tensor Data Mover: the block's 8 ntiles x 2 ksteps
// of B fragments are 16KB contiguous in xf -> one 1-D TENSOR_LOAD_TO_LDS
// (D# per CDNA5 ISA sec.8), issued by wave 0, completed
// with s_wait_tensorcnt, published by the workgroup barrier.
// ---------------------------------------------------------------------------
__global__ __launch_bounds__(256) void k_mixer_wmma(const _Float16* __restrict__ xf,
                                                    const float* __restrict__ w_mixer,
                                                    float* __restrict__ y) {
  __shared__ v16h aLds[256];  // 8KB : [mt][s][lane] -> 16 halves
  __shared__ v16h bLds[512];  // 16KB: [ntile_in_blk][s][lane] -> 16 halves
  int g  = blockIdx.x / 288;
  int tb = blockIdx.x % 288;
  int t  = threadIdx.x;

  // ---- TDM: async-copy this block's 16KB of B fragments into LDS ----------
  if (t < 32) {
    unsigned long long gaddr =
        (unsigned long long)(const void*)(xf + (size_t)(g * NT + tb * 8) * 1024);
    unsigned int ldsAddr = (unsigned int)(unsigned long long)(&bLds[0]);
    v4u g0;
    g0[0] = 1u;                                  // count=1, user descriptor
    g0[1] = ldsAddr;                             // lds_addr (bytes)
    g0[2] = (unsigned int)gaddr;                 // global_addr[31:0]
    g0[3] = ((unsigned int)(gaddr >> 32) & 0x01FFFFFFu) | 0x80000000u; // hi25 | type=2
    v8i g1;
    g1[0] = 0x00020000;       // wg_mask=0, data_size=2 (4B), no pad/iterate
    g1[1] = (4096 << 16);     // tensor_dim0[15:0]=4096 in bits[79:64->31:16]
    g1[2] = (1 << 16);        // tensor_dim0[31:16]=0 ; tensor_dim1[15:0]=1
    g1[3] = (4096 << 16);     // tensor_dim1[31:16]=0 ; tile_dim0=4096 dwords
    g1[4] = 0;                // tile_dim1=0 (unused), tile_dim2=0
    g1[5] = 4096;             // tensor_dim0_stride[31:0]
    g1[6] = 0;                // stride0[47:32]=0, tensor_dim1_stride[15:0]=0
    g1[7] = 0;
    v4i gz4 = {0, 0, 0, 0};             // groups 2/3 unused (<=2-D tensor)
    v8i gz8 = {0, 0, 0, 0, 0, 0, 0, 0}; // extra group (this toolchain's 6-arg form)
    __builtin_amdgcn_tensor_load_to_lds(g0, g1, gz4, gz4, gz8, 0);
  }

  // ---- A stage: convert weights to f16 in WMMA A-fragment layout ----------
  {
    int mt = t >> 6, s = (t >> 5) & 1, L = t & 31;
    int nlo = L & 15, hi = L >> 4;
    int o = mt * 16 + nlo;
    const float* wg = w_mixer + (size_t)g * 4096 + (size_t)o * 64 + s * 32;
    _Float16* dst = (_Float16*)&aLds[(mt * 2 + s) * 32 + L];
#pragma unroll
    for (int e = 0; e < 16; ++e) {
      int v = e >> 1, tt = e & 1;
      int K = hi * 16 + v * 2 + tt;  // A layout K map
      dst[e] = (_Float16)wg[K];
    }
  }
  if (t < 32) __builtin_amdgcn_s_wait_tensorcnt(0);
  __syncthreads();

  int wave = t >> 5, lane = t & 31;
  int nt = tb * 8 + wave;
  int hi = lane >> 4, nlo = lane & 15;
  v8f acc0 = {}, acc1 = {}, acc2 = {}, acc3 = {};
#pragma unroll
  for (int s = 0; s < 2; ++s) {
    v16h bfrag = bLds[(wave * 2 + s) * 32 + lane];
    v16h a0 = aLds[(0 * 2 + s) * 32 + lane];
    v16h a1 = aLds[(1 * 2 + s) * 32 + lane];
    v16h a2 = aLds[(2 * 2 + s) * 32 + lane];
    v16h a3 = aLds[(3 * 2 + s) * 32 + lane];
    acc0 = __builtin_amdgcn_wmma_f32_16x16x32_f16(false, a0, false, bfrag, (short)0, acc0, false, false);
    acc1 = __builtin_amdgcn_wmma_f32_16x16x32_f16(false, a1, false, bfrag, (short)0, acc1, false, false);
    acc2 = __builtin_amdgcn_wmma_f32_16x16x32_f16(false, a2, false, bfrag, (short)0, acc2, false, false);
    acc3 = __builtin_amdgcn_wmma_f32_16x16x32_f16(false, a3, false, bfrag, (short)0, acc3, false, false);
  }
  int ncol = nt * 16 + nlo;
  float* yg = y + (size_t)g * 64 * NTOT + ncol;
#pragma unroll
  for (int r = 0; r < 8; ++r) {
    int mrow = hi * 8 + r;                       // C/D layout: M = hi*8 + r
    yg[(size_t)(mrow) * NTOT]       = acc0[r];
    yg[(size_t)(16 + mrow) * NTOT]  = acc1[r];
    yg[(size_t)(32 + mrow) * NTOT]  = acc2[r];
    yg[(size_t)(48 + mrow) * NTOT]  = acc3[r];
  }
}

// ---------------------------------------------------------------------------
// Per-channel mean / inv-std over NTOT elements (biased var, eps=1e-5).
// stats[2c] = mean, stats[2c+1] = rsqrt(var+eps)
// ---------------------------------------------------------------------------
__global__ __launch_bounds__(256) void k_bn_stats(const float* __restrict__ src,
                                                  float* __restrict__ stats) {
  __shared__ float sh[256], sh2[256];
  int ch = blockIdx.x;
  const float* p = src + (size_t)ch * NTOT;
  float s = 0.f, s2 = 0.f;
  for (int i = threadIdx.x; i < NTOT; i += 256) {
    float v = p[i];
    s += v; s2 += v * v;
  }
  sh[threadIdx.x] = s; sh2[threadIdx.x] = s2;
  __syncthreads();
  for (int st = 128; st > 0; st >>= 1) {
    if ((int)threadIdx.x < st) {
      sh[threadIdx.x]  += sh[threadIdx.x + st];
      sh2[threadIdx.x] += sh2[threadIdx.x + st];
    }
    __syncthreads();
  }
  if (threadIdx.x == 0) {
    float mean = sh[0] * (1.0f / NTOT);
    float var  = sh2[0] * (1.0f / NTOT) - mean * mean;
    stats[2 * ch]     = mean;
    stats[2 * ch + 1] = rsqrtf(var + 1e-5f);
  }
}

// ---------------------------------------------------------------------------
// Stage D: normalize mixer channels 0..63 into x_tem, layout [(b*64+c)][sp]
// ---------------------------------------------------------------------------
__global__ __launch_bounds__(256) void k_xtem(const float* __restrict__ y,
                                              const float* __restrict__ stats1,
                                              const float* __restrict__ g_mnorm,
                                              const float* __restrict__ b_mnorm,
                                              float* __restrict__ xtem) {
  int idx = blockIdx.x * 256 + threadIdx.x;
  if (idx >= 64 * NTOT) return;
  int c = idx / NTOT, n = idx % NTOT;
  int b = n / SP, sp = n % SP;
  float v = y[idx];
  float a = (v - stats1[2 * c]) * stats1[2 * c + 1] * g_mnorm[c] + b_mnorm[c];
  xtem[(size_t)(b * 64 + c) * SP + sp] = a;
}

// ---------------------------------------------------------------------------
// Stage E: 4 fused depthwise convs (incl. sheared pairs, done by index math):
//   h1: taps (dr,dc) dr=-5..5, dc=-1..1
//   v1: taps (dr,dc) dr=-1..1, dc=-5..5
//   h2: taps (dr, dq-dr)    [inv_h(conv(h(x)))]
//   w2: taps (dp-dc, dc)    [inv_v(conv(v(x)))]
// One block per (b,c) 48x48 plane, halo 6 in LDS.
// ---------------------------------------------------------------------------
__global__ __launch_bounds__(256) void k_dwconv4(const float* __restrict__ xtem,
                                                 const float* __restrict__ w_h1,
                                                 const float* __restrict__ w_v1,
                                                 const float* __restrict__ w_h2,
                                                 const float* __restrict__ w_v2,
                                                 float* __restrict__ att) {
  __shared__ float tile[60 * 60];
  __shared__ float wts[132];
  int b = blockIdx.x >> 6, c = blockIdx.x & 63;
  int t = threadIdx.x;
  for (int q = t; q < 132; q += 256) {
    int which = q / 33, k = q % 33;
    const float* src = (which == 0) ? w_h1 : (which == 1) ? w_v1
                       : (which == 2) ? w_h2 : w_v2;
    wts[q] = src[c * 33 + k];
  }
  const float* plane = xtem + (size_t)blockIdx.x * SP;
  for (int q = t; q < 3600; q += 256) {
    int ty = q / 60 - 6, tx = q % 60 - 6;
    float v = 0.f;
    if (ty >= 0 && ty < HS && tx >= 0 && tx < HS) v = plane[ty * HS + tx];
    tile[q] = v;
  }
  __syncthreads();
#pragma unroll
  for (int o = 0; o < 9; ++o) {
    int sp = o * 256 + t;
    int r = sp / HS, cc = sp % HS;
    const float* Tc = &tile[(r + 6) * 60 + (cc + 6)];
    float acc = 0.f;
#pragma unroll
    for (int dr = -5; dr <= 5; ++dr)
#pragma unroll
      for (int dc = -1; dc <= 1; ++dc)
        acc += wts[(dr + 5) * 3 + (dc + 1)] * Tc[dr * 60 + dc];
#pragma unroll
    for (int dr = -1; dr <= 1; ++dr)
#pragma unroll
      for (int dc = -5; dc <= 5; ++dc)
        acc += wts[33 + (dr + 1) * 11 + (dc + 5)] * Tc[dr * 60 + dc];
#pragma unroll
    for (int dr = -5; dr <= 5; ++dr)
#pragma unroll
      for (int dq = -1; dq <= 1; ++dq)
        acc += wts[66 + (dr + 5) * 3 + (dq + 1)] * Tc[dr * 60 + (dq - dr)];
#pragma unroll
    for (int dc = -5; dc <= 5; ++dc)
#pragma unroll
      for (int dp = -1; dp <= 1; ++dp)
        acc += wts[99 + (dp + 1) * 11 + (dc + 5)] * Tc[(dp - dc) * 60 + dc];
    att[(size_t)c * NTOT + b * SP + sp] = acc;
  }
}

// ---------------------------------------------------------------------------
// Stage G: bn2 / bn1, pixel_shuffle3 indexing, sigmoid gate.
// ff channel m = c_out*9 + (y%3)*3 + (x%3); m<64 -> att_map, else mixed ch m.
// ---------------------------------------------------------------------------
__global__ __launch_bounds__(256) void k_final(const float* __restrict__ x,
                                               const float* __restrict__ y,
                                               const float* __restrict__ att,
                                               const float* __restrict__ stats1,
                                               const float* __restrict__ stats2,
                                               const float* __restrict__ g_mnorm,
                                               const float* __restrict__ b_mnorm,
                                               const float* __restrict__ g_norm,
                                               const float* __restrict__ b_norm,
                                               float* __restrict__ out) {
  int idx = blockIdx.x * 256 + threadIdx.x;
  if (idx >= BB * CC * PIX) return;
  int p  = idx % PIX;
  int bc = idx / PIX;
  int co = bc % CC, b = bc / CC;
  int yy = p / WW, xx = p % WW;
  int m = co * 9 + (yy % 3) * 3 + (xx % 3);
  int n = b * SP + (yy / 3) * HS + (xx / 3);
  float a;
  if (m < 64)
    a = (att[(size_t)m * NTOT + n] - stats2[2 * m]) * stats2[2 * m + 1] * g_norm[m] + b_norm[m];
  else
    a = (y[(size_t)m * NTOT + n] - stats1[2 * m]) * stats1[2 * m + 1] * g_mnorm[m] + b_mnorm[m];
  float sg = 1.0f / (1.0f + expf(-a));
  out[idx] = x[idx] * sg;
}

// ---------------------------------------------------------------------------
extern "C" void kernel_launch(void* const* d_in, const int* in_sizes, int n_in,
                              void* d_out, int out_size, void* d_ws, size_t ws_size,
                              hipStream_t stream) {
  const float* x       = (const float*)d_in[0];
  const float* w_mixer = (const float*)d_in[1];
  const float* g_mnorm = (const float*)d_in[2];
  const float* b_mnorm = (const float*)d_in[3];
  const float* w_h1    = (const float*)d_in[4];
  const float* w_v1    = (const float*)d_in[5];
  const float* w_h2    = (const float*)d_in[6];
  const float* w_v2    = (const float*)d_in[7];
  const float* g_norm  = (const float*)d_in[8];
  const float* b_norm  = (const float*)d_in[9];
  float* out = (float*)d_out;

  char* ws = (char*)d_ws;
  size_t off = 0;
  auto carve = [&](size_t bytes) -> char* {
    char* p = ws + off;
    off = (off + bytes + 255) & ~(size_t)255;
    return p;
  };
  _Float16* xf   = (_Float16*)carve((size_t)9 * 64 * NTOT * sizeof(_Float16)); // 42.5 MB
  float* y       = (float*)carve((size_t)576 * NTOT * sizeof(float));          // 84.9 MB
  float* stats1  = (float*)carve((size_t)576 * 2 * sizeof(float));
  float* xtem    = (float*)carve((size_t)BB * 64 * SP * sizeof(float));        // 9.4 MB
  float* attpre  = (float*)carve((size_t)64 * NTOT * sizeof(float));           // 9.4 MB
  float* stats2  = (float*)carve((size_t)64 * 2 * sizeof(float));
  (void)ws_size; (void)in_sizes; (void)n_in; (void)out_size;

  const int nElems = BB * CC * PIX;  // 21,233,664

  k_pool_focus<<<nElems / 256, 256, 0, stream>>>(x, xf);
  k_mixer_wmma<<<9 * 288, 256, 0, stream>>>(xf, w_mixer, y);
  k_bn_stats<<<576, 256, 0, stream>>>(y, stats1);
  k_xtem<<<(64 * NTOT) / 256, 256, 0, stream>>>(y, stats1, g_mnorm, b_mnorm, xtem);
  k_dwconv4<<<BB * 64, 256, 0, stream>>>(xtem, w_h1, w_v1, w_h2, w_v2, attpre);
  k_bn_stats<<<64, 256, 0, stream>>>(attpre, stats2);
  k_final<<<nElems / 256, 256, 0, stream>>>(x, y, attpre, stats1, stats2,
                                            g_mnorm, b_mnorm, g_norm, b_norm, out);
}
